// BDFM_Multi_46428596469850
// MI455X (gfx1250) — matrix operand order repaired and checked by reading, not compiled, not run on hardware
//
#include <hip/hip_runtime.h>
#include <hip/hip_bf16.h>
#include <stdint.h>

// ---------------------------------------------------------------------------
// Problem constants (match reference)
// ---------------------------------------------------------------------------
#define BB   8
#define CC   256
#define HH   128
#define WW   128
#define HWP  (HH * WW)         // 16384
#define NCLS 4
#define K3N  12                // 3*N
#define EPSB 1e-5f

typedef __attribute__((ext_vector_type(8)))  float  v8f;
typedef __attribute__((ext_vector_type(16))) __bf16 v16bf;
typedef __bf16 bf16_t;

// RNE float -> bf16
__device__ __forceinline__ bf16_t f2bf(float f) {
    unsigned u = __builtin_bit_cast(unsigned, f);
    unsigned r = u + 0x7FFFu + ((u >> 16) & 1u);
    unsigned short h = (unsigned short)(r >> 16);
    return __builtin_bit_cast(bf16_t, h);
}

__device__ __forceinline__ v8f wmma_bf16(v16bf a, v16bf b, v8f c) {
    // D = A(16x32 bf16) * B(32x16 bf16) + C(16x16 f32)
    return __builtin_amdgcn_wmma_f32_16x16x32_bf16(
        false, a, false, b, (short)0, c, false, false);
}

// ---------------------------------------------------------------------------
// K0a/K0b: pack W_f (256x256) and W_o (256x512) into bf16 A-fragment order.
// A-fragment order used by this code (16x32): lane L -> row M = mt*16 + (L&15),
// element i -> K = kt*32 + (L>>4)*16 + i (contiguous, matches B reads).
// ---------------------------------------------------------------------------
__global__ void pack_wf_kernel(const float* __restrict__ W, bf16_t* __restrict__ dst) {
    int t = blockIdx.x * 256 + threadIdx.x;          // 8 kt * 16 mt * 32 lanes
    if (t >= 8 * 16 * 32) return;
    int lane = t & 31, mt = (t >> 5) & 15, kt = t >> 9;
    int g = lane >> 4, row = mt * 16 + (lane & 15);
#pragma unroll
    for (int i = 0; i < 16; ++i) {
        int kk = kt * 32 + g * 16 + i;
        dst[(size_t)t * 16 + i] = f2bf(W[row * 256 + kk]);
    }
}

__global__ void pack_wo_kernel(const float* __restrict__ W, bf16_t* __restrict__ dst) {
    int t = blockIdx.x * 256 + threadIdx.x;          // 16 kt * 16 mt * 32 lanes
    if (t >= 16 * 16 * 32) return;
    int lane = t & 31, mt = (t >> 5) & 15, kt = t >> 9;
    int g = lane >> 4, row = mt * 16 + (lane & 15);
#pragma unroll
    for (int i = 0; i < 16; ++i) {
        int kk = kt * 32 + g * 16 + i;
        dst[(size_t)t * 16 + i] = f2bf(W[row * 512 + kk]);   // W_o is 256x512
    }
}

// ---------------------------------------------------------------------------
// K0c: BN inference params: [scale(256) | bias(256)]
// ---------------------------------------------------------------------------
__global__ void bn_prep_kernel(const float* gf, const float* bf, const float* muf, const float* vf,
                               const float* go, const float* bo, const float* muo, const float* vo,
                               float* bnf, float* bno) {
    int c = threadIdx.x;
    float sf = gf[c] * rsqrtf(vf[c] + EPSB);
    bnf[c] = sf;  bnf[256 + c] = bf[c] - muf[c] * sf;
    float so = go[c] * rsqrtf(vo[c] + EPSB);
    bno[c] = so;  bno[256 + c] = bo[c] - muo[c] * so;
}

// ---------------------------------------------------------------------------
// K1: morphology. 4 iters of 4x4 erode (pad 2,1) == min over offsets [-8,+4]^2;
// 4 iters of dilate (pad 1,2) == max over [-4,+8]^2. OOB padding is identity.
// ---------------------------------------------------------------------------
__global__ void morph_row_kernel(const float* __restrict__ m,
                                 float* __restrict__ rmin, float* __restrict__ rmax) {
    int idx = blockIdx.x * 256 + threadIdx.x;        // B*N*H*W = 524288
    if (idx >= BB * NCLS * HWP) return;
    int x = idx & (WW - 1);
    int rowbase = idx & ~(WW - 1);
    float mn = INFINITY, mx = -INFINITY;
#pragma unroll
    for (int dx = -8; dx <= 4; ++dx) {
        int xx = x + dx;
        if ((unsigned)xx < (unsigned)WW) {
            float z = (m[rowbase + xx] > 0.3f) ? 1.0f : 0.0f;
            mn = fminf(mn, z);
        }
    }
#pragma unroll
    for (int dx = -4; dx <= 8; ++dx) {
        int xx = x + dx;
        if ((unsigned)xx < (unsigned)WW) {
            float z = (m[rowbase + xx] > 0.3f) ? 1.0f : 0.0f;
            mx = fmaxf(mx, z);
        }
    }
    rmin[idx] = mn;  rmax[idx] = mx;
}

__global__ void morph_col_kernel(const float* __restrict__ rmin, const float* __restrict__ rmax,
                                 float* __restrict__ er, float* __restrict__ dl) {
    int idx = blockIdx.x * 256 + threadIdx.x;
    if (idx >= BB * NCLS * HWP) return;
    int x = idx & (WW - 1);
    int y = (idx >> 7) & (HH - 1);
    int img = idx >> 14;                             // b*N + cls
    int base = (img << 14) + x;
    float mn = INFINITY, mx = -INFINITY;
#pragma unroll
    for (int dy = -8; dy <= 4; ++dy) {
        int yy = y + dy;
        if ((unsigned)yy < (unsigned)HH) mn = fminf(mn, rmin[base + (yy << 7)]);
    }
#pragma unroll
    for (int dy = -4; dy <= 8; ++dy) {
        int yy = y + dy;
        if ((unsigned)yy < (unsigned)HH) mx = fmaxf(mx, rmax[base + (yy << 7)]);
    }
    er[idx] = mn;  dl[idx] = mx;
}

// ---------------------------------------------------------------------------
// K2: mid[b,k,c] = sum_p fbu[b,k,p] * feature[b,c,p].
// Also (optionally) emits a bf16 tile-contiguous copy of feature:
//   featb[b][tile(256)][p(64)][c(256)]  -- 32 KB per tile, the exact LDS
//   image the fused kernel needs, enabling async global->LDS DMA staging.
// ---------------------------------------------------------------------------
__global__ void zero_mid_kernel(float* mid) {
    int idx = blockIdx.x * 256 + threadIdx.x;
    if (idx < BB * K3N * CC) mid[idx] = 0.0f;
}

#define CHUNK 1024
__global__ void mid_accum_kernel(const float* __restrict__ feature,
                                 const float* __restrict__ er, const float* __restrict__ dl,
                                 float* __restrict__ mid,
                                 bf16_t* __restrict__ featb, int do_pack) {
    __shared__ float sWgt[K3N * CHUNK];              // 48 KB
    int b = blockIdx.x >> 4;                         // 16 chunks per batch
    int p0 = (blockIdx.x & 15) * CHUNK;
    int tid = threadIdx.x;

    for (int it = 0; it < (K3N * CHUNK) / 256; ++it) {
        int e = it * 256 + tid;
        int k = e >> 10, p = e & (CHUNK - 1);
        int cls = k / 3, j = k - cls * 3;
        int mi = ((b * NCLS + cls) << 14) + p0 + p;
        float ev = er[mi], dv = dl[mi];
        float w = (j == 0) ? ev : ((j == 1) ? (1.0f - dv) : (dv - ev));
        sWgt[k * CHUNK + p] = w;
    }
    __syncthreads();

    float acc[K3N];
#pragma unroll
    for (int k = 0; k < K3N; ++k) acc[k] = 0.0f;

    const float* fp = feature + (((size_t)(b * CC + tid)) << 14) + p0;
    for (int p = 0; p < CHUNK; p += 4) {
        float4 f = *(const float4*)(fp + p);
        if (do_pack) {
            int pg = p0 + p;                         // p%4==0 so 4 px share a tile
            int t = pg >> 6, pl = pg & 63;
            bf16_t* dst = featb + ((size_t)((b * 256 + t) * 64 + pl)) * 256 + tid;
            dst[0]   = f2bf(f.x);
            dst[256] = f2bf(f.y);
            dst[512] = f2bf(f.z);
            dst[768] = f2bf(f.w);
        }
#pragma unroll
        for (int k = 0; k < K3N; ++k) {
            const float* w = &sWgt[k * CHUNK + p];
            acc[k] += f.x * w[0] + f.y * w[1] + f.z * w[2] + f.w * w[3];
        }
    }
#pragma unroll
    for (int k = 0; k < K3N; ++k)
        atomicAdd(&mid[b * (K3N * CC) + k * CC + tid], acc[k]);
}

// ---------------------------------------------------------------------------
// K3: fused main kernel (templated on staging mode).
//   USE_ASYNC=1: DMA feature tile (already bf16, tile-contiguous) into LDS
//                with global_load_async_to_lds_b128 (ASYNCcnt).
//   USE_ASYNC=0: float4 global loads + convert + ds_store (fallback).
// Then: GEMM1 -> BN_f -> mid1 -> mid2 -> GEMM4 -> BN_o -> out (WMMA bf16).
// LDS layout (dynamic, 150528 B):
//   sF   bf16 [64][256]   @ 0        sM2  bf16 [64][256] @ 32768
//   sCF  f32  [256][64]   @ 65536    sMid f32  [12][256] @ 131072
//   sM1  f32  [12][64]    @ 143360   sBNf f32  [512]     @ 146432
//   sBNo f32  [512]       @ 148480
// ---------------------------------------------------------------------------
#define SMEM_MAIN 150528

template <int USE_ASYNC>
__global__ void fused_main_kernel(const float* __restrict__ feature,
                                  const bf16_t* __restrict__ featb,
                                  const bf16_t* __restrict__ wfp,
                                  const bf16_t* __restrict__ wop,
                                  const float* __restrict__ mid,
                                  const float* __restrict__ bnf,
                                  const float* __restrict__ bno,
                                  float* __restrict__ out) {
    extern __shared__ float smemf[];
    char*   smem = (char*)smemf;
    bf16_t* sF   = (bf16_t*)smem;
    bf16_t* sM2  = (bf16_t*)(smem + 32768);
    float*  sCF  = (float*)(smem + 65536);
    float*  sMid = (float*)(smem + 131072);
    float*  sM1  = (float*)(smem + 143360);
    float*  sBNf = (float*)(smem + 146432);
    float*  sBNo = (float*)(smem + 148480);

    int b   = blockIdx.x >> 8;                       // 256 tiles per batch
    int p0  = (blockIdx.x & 255) * 64;
    int tid = threadIdx.x;
    int wave = tid >> 5, lane = tid & 31;
    int g = lane >> 4, lm = lane & 15;

    __builtin_prefetch(wfp, 0, 0);                   // global_prefetch_b8

    // --- stage feature tile -> sF[p][c] bf16 ---
    if constexpr (USE_ASYNC) {
        // Tile is a linear 32 KB block: DMA straight into LDS (ASYNCcnt).
        const char* src = (const char*)featb + (size_t)blockIdx.x * 32768;
        uint32_t lbase = (uint32_t)(uintptr_t)sF;    // LDS byte offset (addr[31:0])
#pragma unroll
        for (int i = 0; i < 8; ++i) {
            uint32_t off   = (uint32_t)(tid + i * 256) * 16u;
            uint32_t laddr = lbase + off;
            asm volatile("global_load_async_to_lds_b128 %0, %1, %2 offset:0"
                         :: "v"(laddr), "v"(off), "s"(src)
                         : "memory");
        }
    } else {
        int pl4 = (tid & 15) * 4, cb = tid >> 4;     // 16 c per pass
        const float* fb = feature + ((size_t)b * CC) * HWP + p0;
#pragma unroll
        for (int it = 0; it < 16; ++it) {
            int c = cb + it * 16;
            float4 f = *(const float4*)(fb + (size_t)c * HWP + pl4);
            sF[(pl4 + 0) * 256 + c] = f2bf(f.x);
            sF[(pl4 + 1) * 256 + c] = f2bf(f.y);
            sF[(pl4 + 2) * 256 + c] = f2bf(f.z);
            sF[(pl4 + 3) * 256 + c] = f2bf(f.w);
        }
    }

    // --- stage BN params + mid (overlaps with async DMA) ---
    sBNf[tid] = bnf[tid];           sBNf[tid + 256] = bnf[tid + 256];
    sBNo[tid] = bno[tid];           sBNo[tid + 256] = bno[tid + 256];
#pragma unroll
    for (int it = 0; it < K3N; ++it)
        sMid[it * 256 + tid] = mid[b * (K3N * CC) + it * 256 + tid];

    if constexpr (USE_ASYNC) {
        asm volatile("s_wait_asynccnt 0x0" ::: "memory");
    }
    __syncthreads();

    // =========================== GEMM1: W_f x F ============================
    v8f acc0[4], acc1[4];
#pragma unroll
    for (int nt = 0; nt < 4; ++nt) {
        acc0[nt] = (v8f){0.f, 0.f, 0.f, 0.f, 0.f, 0.f, 0.f, 0.f};
        acc1[nt] = (v8f){0.f, 0.f, 0.f, 0.f, 0.f, 0.f, 0.f, 0.f};
    }
#pragma unroll
    for (int kt = 0; kt < 8; ++kt) {
        v16bf a0 = *(const v16bf*)(wfp + ((size_t)((kt * 16 + wave * 2 + 0) * 32 + lane)) * 16);
        v16bf a1 = *(const v16bf*)(wfp + ((size_t)((kt * 16 + wave * 2 + 1) * 32 + lane)) * 16);
#pragma unroll
        for (int nt = 0; nt < 4; ++nt) {
            // B frag: col = nt*16+lm, K = kt*32 + g*16 + [0..15] (contiguous)
            v16bf bb = *(const v16bf*)(sF + (nt * 16 + lm) * 256 + kt * 32 + g * 16);
            acc0[nt] = wmma_bf16(a0, bb, acc0[nt]);
            acc1[nt] = wmma_bf16(a1, bb, acc1[nt]);
        }
    }
    // BN_f, write CF to LDS (f32 [c][64])
#pragma unroll
    for (int mt = 0; mt < 2; ++mt) {
#pragma unroll
        for (int nt = 0; nt < 4; ++nt) {
#pragma unroll
            for (int r = 0; r < 8; ++r) {
                int mrow = wave * 32 + mt * 16 + g * 8 + r;
                int n = nt * 16 + lm;
                float v = (mt == 0) ? acc0[nt][r] : acc1[nt][r];
                sCF[mrow * 64 + n] = v * sBNf[mrow] + sBNf[256 + mrow];
            }
        }
    }
    __syncthreads();

    // ================== mid1[k,n] = sum_c Mid[k,c] * CF[c,n] ===============
#pragma unroll
    for (int j = 0; j < 3; ++j) {
        int e = tid + j * 256;                       // 768 = 12*64 entries
        int k = e >> 6, n = e & 63;
        float s = 0.0f;
        for (int c = 0; c < 256; ++c)
            s += sMid[k * 256 + c] * sCF[c * 64 + n];
        sM1[e] = s;
    }
    __syncthreads();

    // ========== mid2[c,n] = sum_k Mid[k,c] * mid1[k,n] -> bf16 [n][c] ======
    {
        int c = tid;
        float mk[K3N];
#pragma unroll
        for (int k = 0; k < K3N; ++k) mk[k] = sMid[k * 256 + c];
        for (int n = 0; n < 64; ++n) {
            float s = 0.0f;
#pragma unroll
            for (int k = 0; k < K3N; ++k) s += mk[k] * sM1[k * 64 + n];
            sM2[n * 256 + c] = f2bf(s);
        }
    }
    __syncthreads();

    // ==================== GEMM4: W_o x [F ; mid2], K=512 ====================
#pragma unroll
    for (int nt = 0; nt < 4; ++nt) {
        acc0[nt] = (v8f){0.f, 0.f, 0.f, 0.f, 0.f, 0.f, 0.f, 0.f};
        acc1[nt] = (v8f){0.f, 0.f, 0.f, 0.f, 0.f, 0.f, 0.f, 0.f};
    }
#pragma unroll
    for (int kt = 0; kt < 16; ++kt) {
        const bf16_t* src = (kt < 8) ? sF : sM2;
        int ktl = kt & 7;
        v16bf a0 = *(const v16bf*)(wop + ((size_t)((kt * 16 + wave * 2 + 0) * 32 + lane)) * 16);
        v16bf a1 = *(const v16bf*)(wop + ((size_t)((kt * 16 + wave * 2 + 1) * 32 + lane)) * 16);
#pragma unroll
        for (int nt = 0; nt < 4; ++nt) {
            v16bf bb = *(const v16bf*)(src + (nt * 16 + lm) * 256 + ktl * 32 + g * 16);
            acc0[nt] = wmma_bf16(a0, bb, acc0[nt]);
            acc1[nt] = wmma_bf16(a1, bb, acc1[nt]);
        }
    }
    // BN_o, store out
#pragma unroll
    for (int mt = 0; mt < 2; ++mt) {
#pragma unroll
        for (int nt = 0; nt < 4; ++nt) {
#pragma unroll
            for (int r = 0; r < 8; ++r) {
                int mrow = wave * 32 + mt * 16 + g * 8 + r;
                int n = nt * 16 + lm;
                float v = (mt == 0) ? acc0[nt][r] : acc1[nt][r];
                out[((size_t)(b * CC + mrow)) * HWP + p0 + n] =
                    v * sBNo[mrow] + sBNo[256 + mrow];
            }
        }
    }
}

// ---------------------------------------------------------------------------
// Host launcher
// ---------------------------------------------------------------------------
extern "C" void kernel_launch(void* const* d_in, const int* in_sizes, int n_in,
                              void* d_out, int out_size, void* d_ws, size_t ws_size,
                              hipStream_t stream) {
    const float* feature = (const float*)d_in[0];
    const float* m       = (const float*)d_in[1];
    const float* W_f     = (const float*)d_in[2];
    const float* g_f     = (const float*)d_in[3];
    const float* b_f     = (const float*)d_in[4];
    const float* mu_f    = (const float*)d_in[5];
    const float* v_f     = (const float*)d_in[6];
    const float* W_o     = (const float*)d_in[7];
    const float* g_o     = (const float*)d_in[8];
    const float* b_o     = (const float*)d_in[9];
    const float* mu_o    = (const float*)d_in[10];
    const float* v_o     = (const float*)d_in[11];
    float* out = (float*)d_out;

    char* ws = (char*)d_ws;
    const size_t off_wfp   = 0;                            //  131072 B
    const size_t off_wop   = 131072;                       //  262144 B
    const size_t off_bnf   = 393216;                       //    2048 B
    const size_t off_bno   = 395264;                       //    2048 B
    const size_t off_mid   = 397312;                       //   98304 B
    const size_t off_rmn   = 495616;                       // 2097152 B
    const size_t off_rmx   = 2592768;                      // 2097152 B
    const size_t off_er    = 4689920;                      // 2097152 B
    const size_t off_dl    = 6787072;                      // 2097152 B
    const size_t off_featb = 8884224;                      // 67108864 B (bf16 feature, tiled)
    const size_t ws_need_base  = 8884224;
    const size_t ws_need_async = off_featb + (size_t)BB * HWP * CC * 2;  // 75993088
    if (ws_size < ws_need_base) return;                    // not enough scratch
    const int use_async = (ws_size >= ws_need_async) ? 1 : 0;

    bf16_t* wfp   = (bf16_t*)(ws + off_wfp);
    bf16_t* wop   = (bf16_t*)(ws + off_wop);
    float*  bnf   = (float*)(ws + off_bnf);
    float*  bno   = (float*)(ws + off_bno);
    float*  midp  = (float*)(ws + off_mid);
    float*  rmn   = (float*)(ws + off_rmn);
    float*  rmx   = (float*)(ws + off_rmx);
    float*  erp   = (float*)(ws + off_er);
    float*  dlp   = (float*)(ws + off_dl);
    bf16_t* featb = (bf16_t*)(ws + off_featb);

    // Prep: weight packing (bf16 fragment order) + BN folding
    pack_wf_kernel<<<16, 256, 0, stream>>>(W_f, wfp);
    pack_wo_kernel<<<32, 256, 0, stream>>>(W_o, wop);
    bn_prep_kernel<<<1, 256, 0, stream>>>(g_f, b_f, mu_f, v_f, g_o, b_o, mu_o, v_o, bnf, bno);

    // Morphology (separable 13x13 min/max)
    morph_row_kernel<<<(BB * NCLS * HWP) / 256, 256, 0, stream>>>(m, rmn, rmx);
    morph_col_kernel<<<(BB * NCLS * HWP) / 256, 256, 0, stream>>>(rmn, rmx, erp, dlp);

    // mid = fbu @ feature^T  (+ tiled bf16 feature copy for async staging)
    zero_mid_kernel<<<(BB * K3N * CC + 255) / 256, 256, 0, stream>>>(midp);
    mid_accum_kernel<<<BB * (HWP / CHUNK), 256, 0, stream>>>(feature, erp, dlp, midp,
                                                             featb, use_async);

    // Fused GEMM1 -> mid1 -> mid2 -> GEMM4 -> out
    if (use_async) {
        fused_main_kernel<1><<<BB * (HWP / 64), 256, SMEM_MAIN, stream>>>(
            feature, featb, wfp, wop, midp, bnf, bno, out);
    } else {
        fused_main_kernel<0><<<BB * (HWP / 64), 256, SMEM_MAIN, stream>>>(
            feature, featb, wfp, wop, midp, bnf, bno, out);
    }
}
